// RNNModel_73770358276338
// MI455X (gfx1250) — matrix operand order repaired and checked by reading, not compiled, not run on hardware
//
#include <hip/hip_runtime.h>

// Problem constants (from reference).
#define Bc  8192
#define Tc  8
#define DEc 512
#define DRc 1024
#define Hc  1024

// Vector types (plain ext_vector to avoid HIP struct-vector ctor issues).
typedef __attribute__((ext_vector_type(16))) __bf16    v16bf;
typedef __attribute__((ext_vector_type(8)))  float     v8f;
typedef __attribute__((ext_vector_type(4)))  unsigned  u32x4;
typedef __attribute__((ext_vector_type(2)))  unsigned  u32x2;
typedef __attribute__((ext_vector_type(4)))  float     f32x4;
typedef __attribute__((ext_vector_type(4)))  int       i32x4;

#define BLK 128   // block tile M and N
#define BK  32    // K-step (bf16 elements per LDS row)

// ---------------------------------------------------------------------------
// CDNA5 async global->LDS DMA (ASYNCcnt-tracked), with sync fallback.
// Builtin signature (from hipcc diagnostic): pointers are int4* in AS1/AS3.
// ---------------------------------------------------------------------------
#if __has_builtin(__builtin_amdgcn_global_load_async_to_lds_b128)
#define HAVE_ASYNC_LDS 1
#else
#define HAVE_ASYNC_LDS 0
#endif

__device__ __forceinline__ void cp16_async(const void* g, void* l) {
#if HAVE_ASYNC_LDS
  // Generic->AS casts via integers: LDS generic addresses carry the LDS byte
  // offset in addr[31:0] (ISA 10.2 aperture rules), so truncation is exact.
  __builtin_amdgcn_global_load_async_to_lds_b128(
      (__attribute__((address_space(1))) i32x4*)(unsigned long long)(size_t)g,
      (__attribute__((address_space(3))) i32x4*)(unsigned)(size_t)l,
      0, 0);
#else
  *(u32x4*)l = *(const u32x4*)g;
#endif
}

#if HAVE_ASYNC_LDS
#if __has_builtin(__builtin_amdgcn_s_wait_asynccnt)
#define WAIT_ASYNC(n) __builtin_amdgcn_s_wait_asynccnt(n)
#else
#define WAIT_ASYNC(n) asm volatile("s_wait_asynccnt " #n ::: "memory")
#endif
#else
#define WAIT_ASYNC(n) ((void)0)
#endif

// ---------------------------------------------------------------------------
// fp32 -> bf16 helpers (round to nearest even).
// ---------------------------------------------------------------------------
__device__ __forceinline__ unsigned pack_bf16x2(float a, float b) {
  unsigned ua = __float_as_uint(a), ub = __float_as_uint(b);
  ua = (ua + 0x7FFFu + ((ua >> 16) & 1u)) >> 16;
  ub = (ub + 0x7FFFu + ((ub >> 16) & 1u)) >> 16;
  return (ub << 16) | (ua & 0xFFFFu);
}

__device__ __forceinline__ __bf16 f2bf(float f) {
  unsigned u = __float_as_uint(f);
  u = (u + 0x7FFFu + ((u >> 16) & 1u)) >> 16;
  unsigned short s = (unsigned short)u;
  __bf16 r;
  __builtin_memcpy(&r, &s, 2);
  return r;
}

union FragU {
  u32x4 q[2];
  v16bf v;
};

// ---------------------------------------------------------------------------
// Bulk fp32 -> bf16 conversion (one-time; moves conversion out of GEMM loop).
// ---------------------------------------------------------------------------
__global__ void cvt_f32_bf16_kernel(const float* __restrict__ src,
                                    __bf16* __restrict__ dst, long n4) {
  long i = (long)blockIdx.x * blockDim.x + threadIdx.x;
  long stride = (long)gridDim.x * blockDim.x;
  for (; i < n4; i += stride) {
    f32x4 v = *(const f32x4*)(src + i * 4);
    u32x2 p;
    p.x = pack_bf16x2(v.x, v.y);
    p.y = pack_bf16x2(v.z, v.w);
    *(u32x2*)(dst + i * 4) = p;
  }
}

// ---------------------------------------------------------------------------
// Stage one 128x32 bf16 tile pair (A rows, W rows) into LDS buffers via
// async DMA. Each of 256 threads issues 4 x b128 copies (2 per tile).
// ---------------------------------------------------------------------------
__device__ __forceinline__ void stage_tiles(
    const __bf16* __restrict__ A, size_t lda,
    const __bf16* __restrict__ W, size_t ldw,
    int m0, int n0, int k0, int tid,
    __bf16 (*As)[BK], __bf16 (*Bs)[BK])
{
#pragma unroll
  for (int i = 0; i < 2; ++i) {
    int c   = i * 256 + tid;   // 512 16-byte chunks per tile
    int row = c >> 2;          // 4 chunks (64B) per row
    int qb  = (c & 3) * 16;    // byte offset within row
    const char* ga = (const char*)(A + (size_t)(m0 + row) * lda + k0) + qb;
    cp16_async(ga, (char*)(&As[row][0]) + qb);
    const char* gw = (const char*)(W + (size_t)(n0 + row) * ldw + k0) + qb;
    cp16_async(gw, (char*)(&Bs[row][0]) + qb);
  }
}

// ---------------------------------------------------------------------------
// Double-buffered bf16 WMMA GEMM core: acc += A[M,K] * W[N,K]^T for the
// 128x128 block at (m0,n0). 8 wave32s; each wave owns a 32x64 sub-tile
// (2x4 fragments), 8 v_wmma_f32_16x16x32_bf16 per K-step.
// ---------------------------------------------------------------------------
__device__ __forceinline__ void gemm_bf16(
    const __bf16* __restrict__ A, size_t lda,
    const __bf16* __restrict__ W, size_t ldw,
    int K, int m0, int n0,
    __bf16 (&As)[2][BLK][BK], __bf16 (&Bs)[2][BLK][BK],
    v8f (&acc)[2][4])
{
  const int tid    = threadIdx.x;
  const int lane   = tid & 31;
  const int w      = tid >> 5;
  const int wm0    = (w >> 1) * 32;  // 4 waves along M
  const int wn0    = (w & 1) * 64;   // 2 waves along N
  const int lane15 = lane & 15;
  const int half   = lane >> 4;      // lane half selects K-chunks (ISA layout)

  stage_tiles(A, lda, W, ldw, m0, n0, 0, tid, As[0], Bs[0]);

  int buf = 0;
  for (int k0 = 0; k0 < K; k0 += BK) {
    if (k0 + BK < K) {
      stage_tiles(A, lda, W, ldw, m0, n0, k0 + BK, tid, As[buf ^ 1], Bs[buf ^ 1]);
      WAIT_ASYNC(4);     // current buffer's 4 DMAs retired (in-order)
    } else {
      WAIT_ASYNC(0);
    }
    __syncthreads();     // all waves' DMAs for current buffer landed

    // A fragment (16x32 bf16): per lane, K in [half*8,+8) and [16+half*8,+8).
    FragU af[2];
#pragma unroll
    for (int mf = 0; mf < 2; ++mf) {
      int m = wm0 + mf * 16 + lane15;
      const u32x4* pa = (const u32x4*)&As[buf][m][0];
      af[mf].q[0] = pa[half];
      af[mf].q[1] = pa[2 + half];
    }
    // B fragment (32x16 bf16): per lane, column n, K in [half*8,+8) and [16+half*8,+8).
    FragU bfr[4];
#pragma unroll
    for (int nf = 0; nf < 4; ++nf) {
      int n = wn0 + nf * 16 + lane15;
      const u32x4* pb = (const u32x4*)&Bs[buf][n][0];
      bfr[nf].q[0] = pb[half];
      bfr[nf].q[1] = pb[2 + half];
    }
#pragma unroll
    for (int mf = 0; mf < 2; ++mf)
#pragma unroll
      for (int nf = 0; nf < 4; ++nf)
        acc[mf][nf] = __builtin_amdgcn_wmma_f32_16x16x32_bf16(
            false, af[mf].v, false, bfr[nf].v, (short)0, acc[mf][nf],
            false, false);

    __syncthreads();     // buffer free for the DMA issued next iteration
    buf ^= 1;
  }
}

// ---------------------------------------------------------------------------
// Kernel 1: G[r,n] = ent[r,:]·We[n,:] + rel[r,:]·Wr[n,:] + be[n]+br[n]+bh[n]
// ---------------------------------------------------------------------------
__global__ __launch_bounds__(256) void proj_gemm_kernel(
    const __bf16* __restrict__ eb, const __bf16* __restrict__ rb,
    const __bf16* __restrict__ web, const __bf16* __restrict__ wrb,
    const float* __restrict__ be, const float* __restrict__ br,
    const float* __restrict__ bh, float* __restrict__ G)
{
  __shared__ __bf16 As[2][BLK][BK];
  __shared__ __bf16 Bs[2][BLK][BK];
  const int m0 = blockIdx.y * BLK;   // row in [0, B*T)
  const int n0 = blockIdx.x * BLK;   // col in [0, H)

  v8f acc[2][4] = {};
  gemm_bf16(eb, DEc, web, DEc, DEc, m0, n0, As, Bs, acc);
  gemm_bf16(rb, DRc, wrb, DRc, DRc, m0, n0, As, Bs, acc);

  const int tid = threadIdx.x, lane = tid & 31, w = tid >> 5;
  const int wm0 = (w >> 1) * 32, wn0 = (w & 1) * 64;
  const int lane15 = lane & 15, half = lane >> 4;
#pragma unroll
  for (int nf = 0; nf < 4; ++nf) {
    int gn = n0 + wn0 + nf * 16 + lane15;
    float bias = be[gn] + br[gn] + bh[gn];
#pragma unroll
    for (int mf = 0; mf < 2; ++mf) {
#pragma unroll
      for (int i = 0; i < 8; ++i) {
        // C/D layout: VGPR i holds M = i (lanes 0-15) or 8+i (lanes 16-31).
        int gm = m0 + wm0 + mf * 16 + i + 8 * half;
        G[(size_t)gm * Hc + gn] = acc[mf][nf][i] + bias;
      }
    }
  }
}

// ---------------------------------------------------------------------------
// Kernel 2: h_0 = relu(G_0) (h_{-1}=0, bh already folded into G); writes
// fp32 h_0 to d_out and bf16 h_0 to the recurrent bf16 buffer.
// ---------------------------------------------------------------------------
__global__ void step0_relu_kernel(const float* __restrict__ G,
                                  float* __restrict__ out,
                                  __bf16* __restrict__ hb)
{
  size_t idx = (size_t)blockIdx.x * blockDim.x + threadIdx.x;  // over B*H/4
  size_t b = idx / (Hc / 4);
  size_t c = idx % (Hc / 4);
  size_t off = (size_t)b * Tc * Hc + c * 4;
  f32x4 g = *(const f32x4*)(G + off);
  f32x4 r;
  r.x = g.x > 0.f ? g.x : 0.f;
  r.y = g.y > 0.f ? g.y : 0.f;
  r.z = g.z > 0.f ? g.z : 0.f;
  r.w = g.w > 0.f ? g.w : 0.f;
  *(f32x4*)(out + off) = r;
  u32x2 p;
  p.x = pack_bf16x2(r.x, r.y);
  p.y = pack_bf16x2(r.z, r.w);
  *(u32x2*)(hb + off) = p;
}

// ---------------------------------------------------------------------------
// Kernel 3 (x7): h_t = relu(G_t + h_{t-1}·Wh^T). Reads bf16 h_{t-1}, writes
// fp32 h_t (output) and bf16 h_t (next step's A operand).
// ---------------------------------------------------------------------------
__global__ __launch_bounds__(256) void rnn_step_kernel(
    const __bf16* __restrict__ hb,   // [B,T,H] bf16 hidden states
    const __bf16* __restrict__ whb,  // [H,H] bf16
    const float* __restrict__ G,
    float* __restrict__ out,
    __bf16* __restrict__ hb_out, int t)
{
  __shared__ __bf16 As[2][BLK][BK];
  __shared__ __bf16 Bs[2][BLK][BK];
  const int m0 = blockIdx.y * BLK;   // batch base
  const int n0 = blockIdx.x * BLK;

  v8f acc[2][4] = {};
  gemm_bf16(hb + (size_t)(t - 1) * Hc, (size_t)Tc * Hc,
            whb, Hc, Hc, m0, n0, As, Bs, acc);

  const int tid = threadIdx.x, lane = tid & 31, w = tid >> 5;
  const int wm0 = (w >> 1) * 32, wn0 = (w & 1) * 64;
  const int lane15 = lane & 15, half = lane >> 4;
#pragma unroll
  for (int nf = 0; nf < 4; ++nf) {
    int gn = n0 + wn0 + nf * 16 + lane15;
#pragma unroll
    for (int mf = 0; mf < 2; ++mf) {
#pragma unroll
      for (int i = 0; i < 8; ++i) {
        int gb = m0 + wm0 + mf * 16 + i + 8 * half;
        size_t off = ((size_t)gb * Tc + t) * Hc + gn;
        float v = acc[mf][nf][i] + G[off];
        v = v > 0.f ? v : 0.f;
        out[off] = v;
        hb_out[off] = f2bf(v);
      }
    }
  }
}

// ---------------------------------------------------------------------------
// Kernel 4: need[h][b] = out[b][T-1][h]  (32x32 LDS tile transpose).
// ---------------------------------------------------------------------------
__global__ __launch_bounds__(256) void transpose_need_kernel(
    const float* __restrict__ out, float* __restrict__ need)
{
  __shared__ float tile[32][33];
  const int tx = threadIdx.x & 31;
  const int ty = threadIdx.x >> 5;   // 0..7
  const int h0 = blockIdx.x * 32;
  const int b0 = blockIdx.y * 32;
#pragma unroll
  for (int j = 0; j < 4; ++j) {
    int b = b0 + ty + j * 8;
    tile[ty + j * 8][tx] = out[((size_t)b * Tc + (Tc - 1)) * Hc + (h0 + tx)];
  }
  __syncthreads();
#pragma unroll
  for (int j = 0; j < 4; ++j) {
    int h = h0 + ty + j * 8;
    need[(size_t)h * Bc + (b0 + tx)] = tile[tx][ty + j * 8];
  }
}

// ---------------------------------------------------------------------------
extern "C" void kernel_launch(void* const* d_in, const int* in_sizes, int n_in,
                              void* d_out, int out_size, void* d_ws, size_t ws_size,
                              hipStream_t stream) {
  const float* entity   = (const float*)d_in[0];  // [B,T,DE]
  const float* relation = (const float*)d_in[1];  // [B,T,DR]
  const float* We       = (const float*)d_in[2];  // [H,DE]
  const float* be       = (const float*)d_in[3];  // [H]
  const float* Wr       = (const float*)d_in[4];  // [H,DR]
  const float* br       = (const float*)d_in[5];  // [H]
  const float* Wh       = (const float*)d_in[6];  // [H,H]
  const float* bh       = (const float*)d_in[7];  // [H]
  float* out  = (float*)d_out;                    // [B,T,H] then [H,B]
  float* need = out + (size_t)Bc * Tc * Hc;

  // Workspace layout (bytes).
  char* ws = (char*)d_ws;
  float*  G   = (float*)ws;                 ws += (size_t)Bc * Tc * Hc * 4;   // 268 MB
  __bf16* eb  = (__bf16*)ws;                ws += (size_t)Bc * Tc * DEc * 2;  //  67 MB
  __bf16* rb  = (__bf16*)ws;                ws += (size_t)Bc * Tc * DRc * 2;  // 134 MB
  __bf16* hb  = (__bf16*)ws;                ws += (size_t)Bc * Tc * Hc * 2;   // 134 MB
  __bf16* web = (__bf16*)ws;                ws += (size_t)Hc * DEc * 2;
  __bf16* wrb = (__bf16*)ws;                ws += (size_t)Hc * DRc * 2;
  __bf16* whb = (__bf16*)ws;                ws += (size_t)Hc * Hc * 2;

  dim3 blk(256);
  // One-time fp32 -> bf16 conversions (keeps GEMM loops pure data movement).
  cvt_f32_bf16_kernel<<<4096, blk, 0, stream>>>(entity,   eb,  (long)Bc * Tc * DEc / 4);
  cvt_f32_bf16_kernel<<<4096, blk, 0, stream>>>(relation, rb,  (long)Bc * Tc * DRc / 4);
  cvt_f32_bf16_kernel<<<512,  blk, 0, stream>>>(We,       web, (long)Hc * DEc / 4);
  cvt_f32_bf16_kernel<<<1024, blk, 0, stream>>>(Wr,       wrb, (long)Hc * DRc / 4);
  cvt_f32_bf16_kernel<<<1024, blk, 0, stream>>>(Wh,       whb, (long)Hc * Hc / 4);

  // G = proj(entity, relation) + (be+br+bh)
  proj_gemm_kernel<<<dim3(Hc / BLK, (Bc * Tc) / BLK), blk, 0, stream>>>(
      eb, rb, web, wrb, be, br, bh, G);
  // h_0 = relu(G_0)
  step0_relu_kernel<<<dim3((Bc * Hc / 4) / 256), blk, 0, stream>>>(G, out, hb);
  // h_t = relu(G_t + h_{t-1} Wh^T), sequential over t
  for (int t = 1; t < Tc; ++t)
    rnn_step_kernel<<<dim3(Hc / BLK, Bc / BLK), blk, 0, stream>>>(
        hb, whb, G, out, hb, t);
  // need = h_{T-1}^T
  transpose_need_kernel<<<dim3(Hc / 32, Bc / 32), blk, 0, stream>>>(out, need);
}
